// MambaBlock_35656818491885
// MI455X (gfx1250) — matrix operand order repaired and checked by reading, not compiled, not run on hardware
//
#include <hip/hip_runtime.h>
#include <hip/hip_bf16.h>
#include <math.h>

// Problem constants (from reference)
#define BB  2
#define LL  1024
#define DM  512
#define DI  1024
#define DTR 32
#define DS  16
#define BL  (BB * LL)   // 2048 tokens

typedef __bf16 bf16;
typedef __attribute__((ext_vector_type(16))) __bf16 v16bf;
typedef __attribute__((ext_vector_type(8)))  float  v8f;

__device__ __forceinline__ bf16 f2bf(float f) {
  // round-to-nearest-even fp32 -> bf16
  union { float f; unsigned u; } v; v.f = f;
  unsigned r = v.u + 0x7FFFu + ((v.u >> 16) & 1u);
  unsigned short hh = (unsigned short)(r >> 16);
  return __builtin_bit_cast(bf16, hh);
}

__device__ __forceinline__ float siluf(float x) { return x / (1.f + expf(-x)); }
__device__ __forceinline__ float softplusf(float x) {
  return (x > 20.f) ? x : log1pf(expf(x));
}

// ---------------------------------------------------------------------------
// Bulk fp32 -> bf16 conversion (used once per operand at production time, so
// the GEMM hot loops never pay conversion VALU).
// ---------------------------------------------------------------------------
__global__ void k_f32_to_bf16(const float* __restrict__ src,
                              bf16* __restrict__ dst, int n)
{
  int i = blockIdx.x * blockDim.x + threadIdx.x;
  if (i < n) dst[i] = f2bf(src[i]);
}

// ---------------------------------------------------------------------------
// C[M,N] = act( A[M,K] (row stride lda)  @  W[N,K]^T (row stride ldw) + bias )
// A and W are bf16; WMMA bf16 with fp32 accumulate; fp32 C out, plus an
// optional auxiliary bf16 copy of C (feeds the next GEMM without a pass).
//
// Block: 128 threads = 4 waves in 2x2; block tile 128(M) x 64(N);
// wave tile 64x32 = 4x2 of 16x16 WMMA (8 v_wmma per K-step).
//
// Data movement (CDNA5 path): bf16 K-tiles (A: 128x32, W: 64x32) are pulled
// into LDS with GLOBAL_LOAD_ASYNC_TO_LDS_B128 (ASYNCcnt), double buffered:
// the next tile's async loads are issued before computing the current tile;
// s_wait_asynccnt 0 + workgroup barrier at the top of each iteration makes
// the landed tile visible to all waves (and guarantees the buffer being
// overwritten is no longer read by anyone). Fragment build is pure ds_load.
//
// LDS (dynamic shared, base offset 0): buffer b at b*15360 B:
//   A tile 128 rows x 80 B (row = 32 bf16 = 64 B, padded to 80 B: every b128
//   chunk 16 B-aligned, banks staggered); W tile at +10240 B, 64 rows x 80 B.
//
// Dims assumed: M % 128 == 0, N % 64 == 0, K % 32 == 0 (true for all GEMMs).
// ---------------------------------------------------------------------------
#define GEMM_LDS_BYTES 30720  // 2 buffers * (128 + 64) rows * 80 B

__global__ void gemm_nt_bf16_wmma(const bf16* __restrict__ A, int lda,
                                  const bf16* __restrict__ W, int ldw,
                                  const float* __restrict__ bias,
                                  float* __restrict__ C, int ldc,
                                  bf16* __restrict__ C16, int ldc16,
                                  int M, int N, int K, int act)
{
  extern __shared__ char smem_raw[];
  bf16* smemh = (bf16*)smem_raw;

  const int tid  = threadIdx.x;
  const int lane = tid & 31;
  const int wave = tid >> 5;                   // 0..3
  const int wm = wave >> 1, wn = wave & 1;     // 2x2 wave grid
  const int lo = lane & 15, hi = lane >> 4;    // lane split per ISA 7.12.2
  const int mBlk = blockIdx.y * 128;
  const int nBlk = blockIdx.x * 64;
  (void)M; (void)N;

  v8f acc[4][2];
#pragma unroll
  for (int r = 0; r < 4; ++r)
#pragma unroll
    for (int c = 0; c < 2; ++c)
#pragma unroll
      for (int i = 0; i < 8; ++i) acc[r][c][i] = 0.f;

  // Issue async loads for one bf16 K-tile into LDS buffer `buf`.
  // 768 b128 chunks (A: 512, W: 256), 6 per thread, EXEC all-ones.
  auto stage = [&](int buf, int kb) {
    const unsigned base = (unsigned)buf * 15360u;
    for (int c = tid; c < 768; c += 128) {
      int isB = (c >= 512) ? 1 : 0;            // uniform per iteration
      int cc  = c - (isB << 9);
      int r   = cc >> 2, seg = cc & 3;         // 4 x b128 per 32-elem row
      const bf16* src = isB ? (W + (size_t)(nBlk + r) * ldw + kb + seg * 8)
                            : (A + (size_t)(mBlk + r) * lda + kb + seg * 8);
      unsigned dst = base + (unsigned)isB * 10240u
                   + (unsigned)r * 80u + (unsigned)seg * 16u;
      asm volatile("global_load_async_to_lds_b128 %0, %1, off"
                   :: "v"(dst), "v"(src) : "memory");
    }
  };

  const int nk = K >> 5;
  stage(0, 0);

  for (int it = 0; it < nk; ++it) {
    // Wait for this wave's outstanding async tile, then sync the workgroup.
    asm volatile("s_wait_asynccnt 0x0" ::: "memory");
    __syncthreads();
    if (it + 1 < nk) stage((it + 1) & 1, (it + 1) << 5);

    const unsigned hb = (unsigned)(it & 1) * 7680u;  // bf16-index buffer base

    // A fragment: 16x32, contiguous k order (matches B) — lanes 0-15 hold K 0-15,
    // lanes 16-31 hold K 16-31, M = lane%16.
    v16bf af[4], bw[2];
#pragma unroll
    for (int r = 0; r < 4; ++r) {
      int m = wm * 64 + r * 16 + lo;
#pragma unroll
      for (int i = 0; i < 16; ++i) {
        int kk = hi * 16 + i;
        af[r][i] = smemh[hb + (unsigned)(m * 40 + kk)];
      }
    }
    // B fragment: 32x16, N = lane%16, K = (lane/16)*16 + i.
#pragma unroll
    for (int c = 0; c < 2; ++c) {
      int n = wn * 32 + c * 16 + lo;
#pragma unroll
      for (int i = 0; i < 16; ++i)
        bw[c][i] = smemh[hb + 5120u + (unsigned)(n * 40 + hi * 16 + i)];
    }

#pragma unroll
    for (int r = 0; r < 4; ++r)
#pragma unroll
      for (int c = 0; c < 2; ++c)
        acc[r][c] = __builtin_amdgcn_wmma_f32_16x16x32_bf16(
            false, af[r], false, bw[c], (short)0, acc[r][c], false, false);
  }

  // C/D layout: lanes 0-15: VGPR v -> M=v; lanes 16-31: M=v+8; N = lane%16.
#pragma unroll
  for (int r = 0; r < 4; ++r)
#pragma unroll
    for (int c = 0; c < 2; ++c) {
      int col = nBlk + wn * 32 + c * 16 + lo;
#pragma unroll
      for (int v = 0; v < 8; ++v) {
        int row = mBlk + wm * 64 + r * 16 + v + hi * 8;
        float val = acc[r][c][v];
        if (bias) val += bias[col];
        if (act == 1) val = softplusf(val);
        C[(size_t)row * ldc + col] = val;
        if (C16) C16[(size_t)row * ldc16 + col] = f2bf(val);
      }
    }
}

// ---------------------------------------------------------------------------
// Depthwise SAME conv1d (K=3, pad=1) over L within each batch + bias + SiLU.
// xz: [BL, 2*DI]; channel d from the first DI columns. Emits xs in fp32
// (consumed by k_state) and bf16 (A operand of the x_proj GEMM).
// ---------------------------------------------------------------------------
__global__ void k_conv_silu(const float* __restrict__ xz,
                            const float* __restrict__ cw,
                            const float* __restrict__ cb,
                            float* __restrict__ xs32,
                            bf16* __restrict__ xs16)
{
  int gid = blockIdx.x * blockDim.x + threadIdx.x;
  if (gid >= BL * DI) return;
  int bl = gid >> 10;            // / DI
  int d  = gid & (DI - 1);
  int l  = bl & (LL - 1);        // position within batch
  size_t row = (size_t)bl * (2 * DI);
  float xm1 = (l > 0)      ? xz[row - 2 * DI + d] : 0.f;
  float x0  =                xz[row + d];
  float xp1 = (l < LL - 1) ? xz[row + 2 * DI + d] : 0.f;
  float a = xm1 * cw[d * 3 + 0] + x0 * cw[d * 3 + 1] + xp1 * cw[d * 3 + 2] + cb[d];
  float s = siluf(a);
  xs32[gid] = s;
  xs16[gid] = f2bf(s);
}

// ---------------------------------------------------------------------------
// Fused SSM state update + 4x4 depthwise state fusion + y reduction + z-gate.
// One thread per (b,l,d): all 16 states live in registers; h read once (64 B),
// hn written once (64 B). HBM-bound: ~256 MB => ~11 us floor at 23.3 TB/s.
// yz emitted directly in bf16 (only consumer is the out-proj GEMM).
// ---------------------------------------------------------------------------
__global__ void k_state(const float* __restrict__ h,
                        const float* __restrict__ xz,
                        const float* __restrict__ xs,
                        const float* __restrict__ dBC,
                        const float* __restrict__ delta,
                        const float* __restrict__ A_log,
                        const float* __restrict__ Dvec,
                        const float* __restrict__ sf1w,
                        const float* __restrict__ sf1b,
                        const float* __restrict__ sf2w,
                        const float* __restrict__ sf2b,
                        const float* __restrict__ alpha,
                        float* __restrict__ hn_out,
                        bf16* __restrict__ yz16)
{
  int gid = blockIdx.x * blockDim.x + threadIdx.x;
  if (gid >= BL * DI) return;
  int bl = gid >> 10;
  int d  = gid & (DI - 1);

  float dlt = delta[gid];                           // [BL, DI]
  float xv  = xs[gid];
  float zv  = xz[(size_t)bl * (2 * DI) + DI + d];   // z half of xz
  float dx  = dlt * xv;

  float Bm[DS], Cm[DS];
#pragma unroll
  for (int s = 0; s < DS; ++s) {
    Bm[s] = dBC[(size_t)bl * 64 + 32 + s];          // broadcast across d -> L2 hit
    Cm[s] = dBC[(size_t)bl * 64 + 48 + s];
  }

  const float* hrow = h + (size_t)gid * DS;
  float hp[DS];
#pragma unroll
  for (int s = 0; s < DS; ++s) {
    float Aa = -expf(A_log[d * DS + s]);
    hp[s] = expf(dlt * Aa) * hrow[s] + dx * Bm[s];
  }

  float w1 = sf1w[d], b1 = sf1b[d], b2 = sf2b[d];
  float w2[9];
#pragma unroll
  for (int k = 0; k < 9; ++k) w2[k] = sf2w[d * 9 + k];
  float a0 = alpha[0], a1 = alpha[1];

  float yacc = 0.f;
  float* ho = hn_out + (size_t)gid * DS;
#pragma unroll
  for (int i = 0; i < 4; ++i) {
#pragma unroll
    for (int j = 0; j < 4; ++j) {
      float c3 = 0.f;
#pragma unroll
      for (int ki = 0; ki < 3; ++ki) {
        int ii = i + ki - 1;
        if (ii < 0 || ii > 3) continue;
#pragma unroll
        for (int kj = 0; kj < 3; ++kj) {
          int jj = j + kj - 1;
          if (jj < 0 || jj > 3) continue;
          c3 += w2[ki * 3 + kj] * hp[ii * 4 + jj];
        }
      }
      int s = i * 4 + j;
      float v = a0 * (w1 * hp[s] + b1) + a1 * (c3 + b2);
      ho[s] = v;
      yacc += v * Cm[s];
    }
  }
  float y = yacc + Dvec[d] * xv;
  yz16[gid] = f2bf(y * siluf(zv));
}

// ---------------------------------------------------------------------------
extern "C" void kernel_launch(void* const* d_in, const int* in_sizes, int n_in,
                              void* d_out, int out_size, void* d_ws, size_t ws_size,
                              hipStream_t stream)
{
  (void)in_sizes; (void)n_in; (void)out_size; (void)ws_size;

  const float* x       = (const float*)d_in[0];
  const float* h       = (const float*)d_in[1];
  const float* in_w    = (const float*)d_in[2];
  const float* conv_w  = (const float*)d_in[3];
  const float* conv_b  = (const float*)d_in[4];
  const float* xproj_w = (const float*)d_in[5];
  const float* dt_w    = (const float*)d_in[6];
  const float* dt_b    = (const float*)d_in[7];
  const float* A_log   = (const float*)d_in[8];
  const float* Dvec    = (const float*)d_in[9];
  const float* sf1w    = (const float*)d_in[10];
  const float* sf1b    = (const float*)d_in[11];
  const float* sf2w    = (const float*)d_in[12];
  const float* sf2b    = (const float*)d_in[13];
  const float* alpha   = (const float*)d_in[14];
  const float* out_w   = (const float*)d_in[15];

  // ---- workspace layout ----
  // fp32 section
  float* ws    = (float*)d_ws;
  float* xz    = ws;                          // BL * 2*DI          (16 MB)
  float* xs32  = xz    + (size_t)BL * 2 * DI; // BL * DI            ( 8 MB)
  float* dBC   = xs32  + (size_t)BL * DI;     // BL * 64            (.5 MB)
  float* delta = dBC   + (size_t)BL * 64;     // BL * DI            ( 8 MB)
  // bf16 section (all element counts are multiples of 8 -> 16 B alignment)
  bf16* bws    = (bf16*)(delta + (size_t)BL * DI);
  bf16* xb     = bws;                             // BL*DM
  bf16* in_w16 = xb     + (size_t)BL * DM;        // 2DI*DM
  bf16* xp_w16 = in_w16 + (size_t)2 * DI * DM;    // 64*DI
  bf16* dt_w16 = xp_w16 + (size_t)64 * DI;        // DI*DTR
  bf16* ou_w16 = dt_w16 + (size_t)DI * DTR;       // DM*DI
  bf16* xs16   = ou_w16 + (size_t)DM * DI;        // BL*DI
  bf16* dBC16  = xs16   + (size_t)BL * DI;        // BL*64
  bf16* yz16   = dBC16  + (size_t)BL * 64;        // BL*DI

  float* out    = (float*)d_out;              // BL * DM
  float* hn_out = out + (size_t)BL * DM;      // BL * DI * DS

  const dim3 gblk(128);
  auto cvt = [&](const float* s, bf16* dptr, int n) {
    k_f32_to_bf16<<<(n + 255) / 256, 256, 0, stream>>>(s, dptr, n);
  };

  // 0) one-time bf16 conversions of GEMM operands (~2.7M elems, negligible)
  cvt(x,       xb,     BL * DM);
  cvt(in_w,    in_w16, 2 * DI * DM);
  cvt(xproj_w, xp_w16, 64 * DI);
  cvt(dt_w,    dt_w16, DI * DTR);
  cvt(out_w,   ou_w16, DM * DI);

  // 1) xz = x @ in_proj_w^T               [2048 x 2048], K=512
  gemm_nt_bf16_wmma<<<dim3((2 * DI) / 64, BL / 128), gblk, GEMM_LDS_BYTES, stream>>>(
      xb, DM, in_w16, DM, nullptr, xz, 2 * DI, nullptr, 0, BL, 2 * DI, DM, 0);

  // 2) depthwise conv1d + bias + SiLU -> xs (fp32 + bf16)
  k_conv_silu<<<(BL * DI + 255) / 256, 256, 0, stream>>>(
      xz, conv_w, conv_b, xs32, xs16);

  // 3) dBC = xs @ x_proj_w^T              [2048 x 64], K=1024 (+ bf16 copy)
  gemm_nt_bf16_wmma<<<dim3(64 / 64, BL / 128), gblk, GEMM_LDS_BYTES, stream>>>(
      xs16, DI, xp_w16, DI, nullptr, dBC, 64, dBC16, 64, BL, 64, DI, 0);

  // 4) delta = softplus(dBC16[:, :32] @ dt_proj_w^T + b)  [2048 x 1024], K=32
  gemm_nt_bf16_wmma<<<dim3(DI / 64, BL / 128), gblk, GEMM_LDS_BYTES, stream>>>(
      dBC16, 64, dt_w16, DTR, dt_b, delta, DI, nullptr, 0, BL, DI, DTR, 1);

  // 5) fused state update + 4x4 fusion + y reduction + z-gate -> yz (bf16)
  k_state<<<(BL * DI + 255) / 256, 256, 0, stream>>>(
      h, xz, xs32, dBC, delta, A_log, Dvec, sf1w, sf1b, sf2w, sf2b, alpha,
      hn_out, yz16);

  // 6) out = yz @ out_proj_w^T            [2048 x 512], K=1024
  gemm_nt_bf16_wmma<<<dim3(DM / 64, BL / 128), gblk, GEMM_LDS_BYTES, stream>>>(
      yz16, DI, ou_w16, DI, nullptr, out, DM, nullptr, 0, BL, DM, DI, 0);
}
